// CrossAttention_18038862643955
// MI455X (gfx1250) — matrix operand order repaired and checked by reading, not compiled, not run on hardware
//
#include <hip/hip_runtime.h>

// ---------------------------------------------------------------------------
// CDNA5 (gfx1250) cross-attention, bf16 WMMA pipeline with fused
// scores+softmax+PV kernel using the 320KB/WGP LDS.
// bs=4, n_q=1024, n_c=4096, D=512, H=8, hd=64, scale=1/8, tau=1.
// ---------------------------------------------------------------------------

typedef __attribute__((ext_vector_type(16))) __bf16 v16bf;
typedef __attribute__((ext_vector_type(8)))  __bf16 v8bf;
typedef __attribute__((ext_vector_type(4)))  __bf16 v4bf;
typedef __attribute__((ext_vector_type(8)))  float  v8f;
typedef __attribute__((ext_vector_type(4)))  float  v4f;

#define D_MODEL 512
#define NHEADS  8
#define HDIM    64
#define NQ      1024
#define NC      4096
#define BSZ     4
#define SCALE   0.125f

// ---- WMMA wrapper ---------------------------------------------------------
__device__ __forceinline__ v8f wmma_bf16(v16bf a, v16bf b, v8f c) {
  return __builtin_amdgcn_wmma_f32_16x16x32_bf16(
      /*neg_a=*/false, a, /*neg_b=*/false, b,
      /*c_mod=*/(short)0, c, /*reuse_a=*/false, /*reuse_b=*/false);
}

// ---- Fragment loaders (contiguous k order, same in A and B) ----------------------
// A (16x32 bf16, M x K): lane m in [0,15] holds row m; lane-half selects K
// sub-bands: elems 0..15 -> K = hl*16 + e (contiguous, matches load_b_bf16).
// Works for global or LDS pointers (generic addressing).
__device__ __forceinline__ v16bf load_a_bf16(const __bf16* base, int ld) {
  int lane = threadIdx.x & 31;
  int m = lane & 15, hl = lane >> 4;
  const __bf16* p = base + (size_t)m * ld;
  v8bf lo = *(const v8bf*)(p + hl * 16);
  v8bf hi = *(const v8bf*)(p + hl * 16 + 8);
  v16bf a;
#pragma unroll
  for (int i = 0; i < 8; ++i) { a[i] = lo[i]; a[8 + i] = hi[i]; }
  return a;
}

// B (32x16 bf16, K x N) built from a row-major [N, K] source (i.e. B = src^T):
// lane n = column; lane-half selects K in [0,16) or [16,32); one 32B load.
__device__ __forceinline__ v16bf load_b_bf16(const __bf16* base, int ld) {
  int lane = threadIdx.x & 31;
  int n = lane & 15, hl = lane >> 4;
  return *(const v16bf*)(base + (size_t)n * ld + hl * 16);
}

// ---- f32 -> bf16 convert --------------------------------------------------
__global__ __launch_bounds__(256) void cvt_bf16(const float* __restrict__ in,
                                                __bf16* __restrict__ out, int n) {
  int i = blockIdx.x * blockDim.x + threadIdx.x;
  if (i < n) out[i] = (__bf16)in[i];
}

// ---- Generic GEMM: Y[M,512] = X[M,512] @ W[512,512]^T + bias --------------
// MODE 0: bf16 row-major out.  MODE 1: bf16 transposed-per-head out
// ([b,h,hd,n_c], for V).      MODE 2: f32 row-major out (final output).
template <int MODE>
__global__ __launch_bounds__(128) void gemm_xwT(const __bf16* __restrict__ X,
                                                const __bf16* __restrict__ W,
                                                const float* __restrict__ bias,
                                                __bf16* __restrict__ outb,
                                                float* __restrict__ outf) {
  const int K = D_MODEL;
  int wave = threadIdx.x >> 5, lane = threadIdx.x & 31;
  int wm = wave & 1, wn = wave >> 1;
  int m0 = blockIdx.y * 64 + wm * 32;
  int n0 = blockIdx.x * 64 + wn * 32;

  v8f acc[2][2] = {};
  for (int k0 = 0; k0 < K; k0 += 32) {
    v16bf a0 = load_a_bf16(X + (size_t)m0 * K + k0, K);
    v16bf a1 = load_a_bf16(X + (size_t)(m0 + 16) * K + k0, K);
    v16bf b0 = load_b_bf16(W + (size_t)n0 * K + k0, K);
    v16bf b1 = load_b_bf16(W + (size_t)(n0 + 16) * K + k0, K);
    acc[0][0] = wmma_bf16(a0, b0, acc[0][0]);
    acc[0][1] = wmma_bf16(a0, b1, acc[0][1]);
    acc[1][0] = wmma_bf16(a1, b0, acc[1][0]);
    acc[1][1] = wmma_bf16(a1, b1, acc[1][1]);
  }

  int cn = lane & 15, ch = lane >> 4;
#pragma unroll
  for (int mi = 0; mi < 2; ++mi) {
#pragma unroll
    for (int ni = 0; ni < 2; ++ni) {
      int col = n0 + ni * 16 + cn;
      int row0 = m0 + mi * 16 + ch * 8;
      float bv = bias[col];
      if constexpr (MODE == 1) {
        // transposed store: v_t[b][h][d][j]; lane-contiguous over j -> 16B store
        int b = row0 >> 12;          // / NC
        int j = row0 & (NC - 1);
        int h = col >> 6, d = col & 63;
        __bf16* p = outb + ((size_t)((b * NHEADS + h) * HDIM + d)) * NC + j;
        v8bf ov;
#pragma unroll
        for (int r = 0; r < 8; ++r) ov[r] = (__bf16)(acc[mi][ni][r] + bv);
        *(v8bf*)p = ov;
      } else {
#pragma unroll
        for (int r = 0; r < 8; ++r) {
          size_t idx = (size_t)(row0 + r) * D_MODEL + col;
          float v = acc[mi][ni][r] + bv;
          if constexpr (MODE == 0) outb[idx] = (__bf16)v;
          else                     outf[idx] = v;
        }
      }
    }
  }
}

// ---------------------------------------------------------------------------
// Fused attention: one workgroup owns 16 query rows x full 4096 keys of one
// (b,h). Scores -> LDS f32 (256KB, needs CDNA5's 320KB WGP LDS) -> softmax
// -> normalized f32 attn written once to d_out + bf16 probs packed in place
// -> PV GEMM (A from LDS, B = transposed V from global) -> ctx bf16.
// Dynamic LDS layout (262272 bytes total):
//   [0, 262144)      : Sf (float[16][4096]); low half aliased as Sb bf16
//   [131072, 139264) : PV split-K partials (float[8][256]) -- phase 4 only
//   [262144, 262272) : row stats m[16], inv[16]
// ---------------------------------------------------------------------------
__global__ __launch_bounds__(256) void fused_attn(const __bf16* __restrict__ Q,
                                                  const __bf16* __restrict__ Kp,
                                                  const __bf16* __restrict__ VT,
                                                  const float* __restrict__ mask,
                                                  float* __restrict__ attn,
                                                  __bf16* __restrict__ CTX) {
  extern __shared__ char smem[];
  float*  Sf   = (float*)smem;
  __bf16* Sb   = (__bf16*)smem;
  float*  red  = (float*)(smem + 131072);
  float*  stat = (float*)(smem + 262144);

  int bh = blockIdx.y, b = bh >> 3, h = bh & 7;
  int i0 = blockIdx.x * 16;
  int tid = threadIdx.x;
  int wave = tid >> 5, lane = tid & 31;
  int cn = lane & 15, ch = lane >> 4;

  const __bf16* qb = Q + (size_t)b * NQ * D_MODEL + (size_t)i0 * D_MODEL + h * HDIM;
  const __bf16* kb = Kp + (size_t)b * NC * D_MODEL + h * HDIM;

  // ---- Phase 1: scores*SCALE + mask -> LDS f32 (wave w owns 512 cols) ----
  v16bf aq0 = load_a_bf16(qb, D_MODEL);        // K = 0..31 of head dim
  v16bf aq1 = load_a_bf16(qb + 32, D_MODEL);   // K = 32..63
  const float* mrow = mask + (size_t)i0 * NC;
  for (int jt = 0; jt < 32; ++jt) {
    int j0 = wave * 512 + jt * 16;
    v16bf b0 = load_b_bf16(kb + (size_t)j0 * D_MODEL, D_MODEL);
    v16bf b1 = load_b_bf16(kb + (size_t)j0 * D_MODEL + 32, D_MODEL);
    v8f acc = {};
    acc = wmma_bf16(aq0, b0, acc);
    acc = wmma_bf16(aq1, b1, acc);
    int col = j0 + cn;
#pragma unroll
    for (int r = 0; r < 8; ++r) {
      int lr = ch * 8 + r;
      Sf[lr * NC + col] = acc[r] * SCALE + mrow[(size_t)lr * NC + col];
    }
  }
  __syncthreads();

  // ---- Phase 2: per-row online softmax stats (wave w -> rows 2w, 2w+1) ----
  for (int rr = wave * 2; rr < wave * 2 + 2; ++rr) {
    float m = -INFINITY, s = 0.f;
    for (int j = lane; j < NC; j += 32) {
      float x = Sf[rr * NC + j];
      float nm = fmaxf(m, x);
      s = s * __expf(m - nm) + __expf(x - nm);
      m = nm;
    }
#pragma unroll
    for (int off = 16; off; off >>= 1) {
      float mo = __shfl_xor(m, off, 32);
      float so = __shfl_xor(s, off, 32);
      float nm = fmaxf(m, mo);
      s = s * __expf(m - nm) + so * __expf(mo - nm);
      m = nm;
    }
    if (lane == 0) { stat[rr] = m; stat[16 + rr] = 1.f / s; }
  }
  __syncthreads();

  // ---- Phase 3: normalize; write f32 attn to global; pack bf16 in place ----
  // In-place 2x compaction: bf16 writes always land below all future f32
  // reads; the per-iteration barrier covers the one overlapping iteration.
  float* arow = attn + ((size_t)bh * NQ + i0) * NC;
  for (int t = 0; t < 64; ++t) {
    int idx4 = t * 256 + tid;          // float4 index into 16x4096
    int row = idx4 >> 10;
    int c4 = (idx4 & 1023) * 4;
    v4f x = ((const v4f*)Sf)[idx4];
    float m = stat[row], inv = stat[16 + row];
    v4f y;
#pragma unroll
    for (int e = 0; e < 4; ++e) y[e] = __expf(x[e] - m) * inv;
    *(v4f*)(arow + (size_t)row * NC + c4) = y;
    __syncthreads();
    v4bf yb;
#pragma unroll
    for (int e = 0; e < 4; ++e) yb[e] = (__bf16)y[e];
    ((v4bf*)Sb)[idx4] = yb;
  }
  __syncthreads();

  // ---- Phase 4: PV. wave w: n-tile (w&3), split-K half (w>>2) ----
  int d0 = (wave & 3) * 16;
  int kbeg = (wave >> 2) * 2048;
  const __bf16* Vb = VT + (size_t)bh * HDIM * NC;
  v8f acc = {};
  for (int k0 = kbeg; k0 < kbeg + 2048; k0 += 32) {
    v16bf a0 = load_a_bf16(Sb + k0, NC);                      // LDS ds loads
    v16bf b0 = load_b_bf16(Vb + (size_t)d0 * NC + k0, NC);    // global 32B loads
    acc = wmma_bf16(a0, b0, acc);
  }
#pragma unroll
  for (int r = 0; r < 8; ++r)
    red[wave * 256 + (ch * 8 + r) * 16 + cn] = acc[r];
  __syncthreads();
  if (wave < 4) {
    int col = h * HDIM + d0 + cn;
#pragma unroll
    for (int r = 0; r < 8; ++r) {
      int lr = ch * 8 + r;
      float v = red[wave * 256 + lr * 16 + cn] + red[(wave + 4) * 256 + lr * 16 + cn];
      CTX[((size_t)b * NQ + i0 + lr) * D_MODEL + col] = (__bf16)v;
    }
  }
}

// ---------------------------------------------------------------------------
extern "C" void kernel_launch(void* const* d_in, const int* in_sizes, int n_in,
                              void* d_out, int out_size, void* d_ws, size_t ws_size,
                              hipStream_t stream) {
  const float* query   = (const float*)d_in[0];
  const float* context = (const float*)d_in[1];
  const float* mask    = (const float*)d_in[2];
  const float* Wq = (const float*)d_in[3];  const float* bq = (const float*)d_in[4];
  const float* Wk = (const float*)d_in[5];  const float* bk = (const float*)d_in[6];
  const float* Wv = (const float*)d_in[7];  const float* bv = (const float*)d_in[8];
  const float* Wo = (const float*)d_in[9];  const float* bo = (const float*)d_in[10];

  float* out  = (float*)d_out;                                   // [4,1024,512]
  float* attn = out + (size_t)BSZ * NQ * D_MODEL;                // [4,8,1024,4096]

  const size_t NQD = (size_t)BSZ * NQ * D_MODEL;   // 2,097,152
  const size_t NCD = (size_t)BSZ * NC * D_MODEL;   // 8,388,608
  const size_t DD  = (size_t)D_MODEL * D_MODEL;    // 262,144

  __bf16* ws = (__bf16*)d_ws;
  size_t o = 0;
  __bf16* qx   = ws + o; o += NQD;   // query bf16
  __bf16* cx   = ws + o; o += NCD;   // context bf16
  __bf16* wqb  = ws + o; o += DD;
  __bf16* wkb  = ws + o; o += DD;
  __bf16* wvb  = ws + o; o += DD;
  __bf16* wob  = ws + o; o += DD;
  __bf16* qp   = ws + o; o += NQD;   // Q projection  [bs*nq, 512]
  __bf16* kp   = ws + o; o += NCD;   // K projection  [bs*nc, 512]
  __bf16* vt   = ws + o; o += NCD;   // V projection transposed [bs,H,hd,nc]
  __bf16* ctxb = ws + o; o += NQD;   // attention context [bs*nq, 512]

  // --- converts ---
  cvt_bf16<<<(int)((NQD + 255) / 256), 256, 0, stream>>>(query,   qx,  (int)NQD);
  cvt_bf16<<<(int)((NCD + 255) / 256), 256, 0, stream>>>(context, cx,  (int)NCD);
  cvt_bf16<<<(int)((DD  + 255) / 256), 256, 0, stream>>>(Wq,      wqb, (int)DD);
  cvt_bf16<<<(int)((DD  + 255) / 256), 256, 0, stream>>>(Wk,      wkb, (int)DD);
  cvt_bf16<<<(int)((DD  + 255) / 256), 256, 0, stream>>>(Wv,      wvb, (int)DD);
  cvt_bf16<<<(int)((DD  + 255) / 256), 256, 0, stream>>>(Wo,      wob, (int)DD);

  dim3 blk(128);
  // --- projections ---
  gemm_xwT<0><<<dim3(8,  64, 1), blk, 0, stream>>>(qx, wqb, bq, qp, nullptr);
  gemm_xwT<0><<<dim3(8, 256, 1), blk, 0, stream>>>(cx, wkb, bk, kp, nullptr);
  gemm_xwT<1><<<dim3(8, 256, 1), blk, 0, stream>>>(cx, wvb, bv, vt, nullptr);

  // --- fused scores + softmax + attn write + PV ---
  const size_t SMEM = 16 * (size_t)NC * 4 + 128;   // 262272 bytes
  fused_attn<<<dim3(NQ / 16, BSZ * NHEADS), 256, SMEM, stream>>>(qp, kp, vt, mask,
                                                                 attn, ctxb);

  // --- output projection (f32 out) ---
  gemm_xwT<2><<<dim3(8, 64, 1), blk, 0, stream>>>(ctxb, wob, bo, nullptr, out);
}
